// DotProductAttention_29549374997028
// MI455X (gfx1250) — matrix operand order repaired and checked by reading, not compiled, not run on hardware
//
#include <hip/hip_runtime.h>

typedef __attribute__((ext_vector_type(16))) _Float16 v16h;
typedef __attribute__((ext_vector_type(8)))  float    v8f;

#define BATCH 32
#define SQ    2048
#define SK    2048
#define DH    128     // head dim
#define BQ    128     // q rows per block
#define BK    64      // keys per inner iteration
#define NWAVE 8

// Raw hardware exp2 (v_exp_f32).
#define EXP2F(x) __builtin_amdgcn_exp2f(x)

// LDS: phase 1 stages the whole Q tile (f16); phase 2 reuses the space for
// K (row-major), V (transposed, d-major) and per-wave P scratch.
union SMem {
  _Float16 qstage[BQ * DH];                 // 32 KB
  struct {
    _Float16 k[BK * DH];                    // 16 KB  k[key][d]
    _Float16 vT[DH * BK];                   // 16 KB  vT[d][key]
    _Float16 p[NWAVE][16 * BK];             // 16 KB  per-wave P scratch
  } s;
};

__device__ __forceinline__ v8f wmma_f16(v16h a, v16h b, v8f c) {
  return __builtin_amdgcn_wmma_f32_16x16x32_f16(
      false, a, false, b, (short)0, c, false, false);
}

// One flash-attention step over BK=64 keys staged in LDS.
template <bool MASKED>
__device__ __forceinline__ void fa_step(
    int kb, int n, int g, int rowbase,
    const _Float16* __restrict__ ksm, const _Float16* __restrict__ vsm,
    _Float16* __restrict__ pw,
    const v16h* __restrict__ qfrag, v16h onesv,
    v8f* __restrict__ acc, v8f& accl, float* __restrict__ mrow)
{
  // ---- S = Q * K^T : four 16x16 f32 tiles (scores already in log2 units) ----
  v8f st[4];
#pragma unroll
  for (int t = 0; t < 4; ++t) {
    v8f c = {};
#pragma unroll
    for (int dk = 0; dk < 4; ++dk) {
      // B-frag 32x16: lane holds key column t*16+n; d runs g*16..g*16+15
      v16h bk = *(const v16h*)(&ksm[(t * 16 + n) * DH + dk * 32 + g * 16]);
      c = wmma_f16(qfrag[dk], bk, c);
    }
    st[t] = c;
  }

  // ---- online softmax (exp2 domain); row r of VGPR <-> q row rowbase+r ----
#pragma unroll
  for (int r = 0; r < 8; ++r) {
    const int qrow = rowbase + r;
    float s0 = st[0][r], s1 = st[1][r], s2 = st[2][r], s3 = st[3][r];
    if (MASKED) {
      if (kb + n      > qrow) s0 = -1e30f;
      if (kb + 16 + n > qrow) s1 = -1e30f;
      if (kb + 32 + n > qrow) s2 = -1e30f;
      if (kb + 48 + n > qrow) s3 = -1e30f;
    }
    float mx = fmaxf(fmaxf(fmaxf(s0, s1), fmaxf(s2, s3)), mrow[r]);
#pragma unroll
    for (int off = 1; off < 16; off <<= 1)
      mx = fmaxf(mx, __shfl_xor(mx, off, 32));
    const float corr = EXP2F(mrow[r] - mx);
    mrow[r] = mx;
    st[0][r] = EXP2F(s0 - mx);
    st[1][r] = EXP2F(s1 - mx);
    st[2][r] = EXP2F(s2 - mx);
    st[3][r] = EXP2F(s3 - mx);
    accl[r] *= corr;
#pragma unroll
    for (int nt = 0; nt < 8; ++nt) acc[nt][r] *= corr;
  }

  // ---- C-layout P -> A-fragment layout via per-wave LDS scratch ----
#pragma unroll
  for (int r = 0; r < 8; ++r) {
    const int row = g * 8 + r;
#pragma unroll
    for (int t = 0; t < 4; ++t)
      pw[row * BK + t * 16 + n] = (_Float16)st[t][r];
  }
  asm volatile("s_wait_dscnt 0" ::: "memory");  // wave-internal LDS RAW

  v16h pfrag[2];
#pragma unroll
  for (int h = 0; h < 2; ++h) {
    const _Float16* prow = &pw[n * BK + h * 32];
    union { v16h v; float4 f4[2]; } u;
    u.f4[0] = *(const float4*)(prow + g * 16);
    u.f4[1] = *(const float4*)(prow + g * 16 + 8);
    pfrag[h] = u.v;
  }

  // ---- softmax denominator via WMMA: accl += P * ones ----
  accl = wmma_f16(pfrag[0], onesv, accl);
  accl = wmma_f16(pfrag[1], onesv, accl);

  // ---- O += P * V : 8 output d-tiles, K-dim 64 = 2 chunks ----
#pragma unroll
  for (int nt = 0; nt < 8; ++nt) {
    v16h bv0 = *(const v16h*)(&vsm[(nt * 16 + n) * BK + g * 16]);
    v16h bv1 = *(const v16h*)(&vsm[(nt * 16 + n) * BK + 32 + g * 16]);
    acc[nt] = wmma_f16(pfrag[0], bv0, acc[nt]);
    acc[nt] = wmma_f16(pfrag[1], bv1, acc[nt]);
  }
}

__global__ __launch_bounds__(256)
void fa_causal_f16_wmma(const float* __restrict__ Q,
                        const float* __restrict__ K,
                        const float* __restrict__ V,
                        float* __restrict__ O)
{
  __shared__ SMem sm;

  const int lane = threadIdx.x & 31;
  const int wid  = threadIdx.x >> 5;
  const int n    = lane & 15;   // column-within-tile for B/C/D layouts
  const int g    = lane >> 4;   // 16-lane group (0/1)

  const int qtiles = SQ / BQ;
  const int b  = blockIdx.x / qtiles;
  const int q0 = (blockIdx.x % qtiles) * BQ;

  // (1/sqrt(128)) * log2(e): scores come out of QK^T in log2 units.
  const float scale_l2e = 0.088388347648318447f * 1.4426950408889634f;

  const float* Qb = Q + ((size_t)b * SQ + q0) * DH;
  const float* Kb = K + (size_t)b * SK * DH;
  const float* Vb = V + (size_t)b * SK * DH;
  float*       Ob = O + ((size_t)b * SQ + q0) * DH;

  // ---------------- stage Q tile (f32 -> pre-scaled f16 LDS) ----------------
  for (int i = threadIdx.x; i < BQ * DH / 4; i += 256) {
    float4 f = ((const float4*)Qb)[i];
    _Float16* d = &sm.qstage[i * 4];
    d[0] = (_Float16)(f.x * scale_l2e); d[1] = (_Float16)(f.y * scale_l2e);
    d[2] = (_Float16)(f.z * scale_l2e); d[3] = (_Float16)(f.w * scale_l2e);
  }
  __syncthreads();

  // Q A-fragments (16x32 f16): lane m=lane%16 holds row m; k runs
  // [g*16 .. g*16+15] (contiguous, matching B fragments).
  v16h qfrag[4];
  {
    const _Float16* qrow = &sm.qstage[(wid * 16 + n) * DH];
#pragma unroll
    for (int dk = 0; dk < 4; ++dk) {
      union { v16h h; float4 f4[2]; } u;
      u.f4[0] = *(const float4*)(qrow + dk * 32 + g * 16);
      u.f4[1] = *(const float4*)(qrow + dk * 32 + g * 16 + 8);
      qfrag[dk] = u.h;
    }
  }
  __syncthreads();

  v16h onesv;
#pragma unroll
  for (int i = 0; i < 16; ++i) onesv[i] = (_Float16)1.0f;

  // ---------------- state ----------------
  v8f acc[8] = {};   // 16 rows x 128 cols f32 output accumulator
  v8f accl = {};     // softmax denominator (P x ones), same C layout
  float mrow[8];
#pragma unroll
  for (int r = 0; r < 8; ++r) mrow[r] = -1e30f;

  const int rowbase = q0 + wid * 16 + g * 8;   // global q row of acc[.][r]

  // stage K (row major) + V (transposed) for tile kb, prefetch tile kb+BK
  auto stageKV = [&](int kb) {
    const float* kp = Kb + (size_t)kb * DH;
    const float* vp = Vb + (size_t)kb * DH;
    const bool pf = (kb + BK) < SK;
    for (int i = threadIdx.x; i < BK * DH / 4; i += 256) {
      float4 fk = ((const float4*)kp)[i];
      float4 fv = ((const float4*)vp)[i];
      if (pf) {
        __builtin_prefetch(kp + BK * DH + i * 4, 0, 1);
        __builtin_prefetch(vp + BK * DH + i * 4, 0, 1);
      }
      _Float16* dk_ = &sm.s.k[i * 4];
      dk_[0] = (_Float16)fk.x; dk_[1] = (_Float16)fk.y;
      dk_[2] = (_Float16)fk.z; dk_[3] = (_Float16)fk.w;
      const int kv = (i * 4) / DH;        // key row
      const int d0 = (i * 4) % DH;        // d column
      sm.s.vT[(d0 + 0) * BK + kv] = (_Float16)fv.x;
      sm.s.vT[(d0 + 1) * BK + kv] = (_Float16)fv.y;
      sm.s.vT[(d0 + 2) * BK + kv] = (_Float16)fv.z;
      sm.s.vT[(d0 + 3) * BK + kv] = (_Float16)fv.w;
    }
  };

  _Float16* pw = sm.s.p[wid];

  // ---- bulk: kb+BK-1 < q0 <= every q row in block -> no masking ----
  int kb = 0;
  for (; kb < q0; kb += BK) {
    stageKV(kb);
    __syncthreads();
    fa_step<false>(kb, n, g, rowbase, sm.s.k, sm.s.vT, pw,
                   qfrag, onesv, acc, accl, mrow);
    __syncthreads();
  }
  // ---- tail: diagonal region, causal masking active (2 iterations) ----
  for (; kb < q0 + BQ; kb += BK) {
    stageKV(kb);
    __syncthreads();
    fa_step<true>(kb, n, g, rowbase, sm.s.k, sm.s.vT, pw,
                  qfrag, onesv, acc, accl, mrow);
    __syncthreads();
  }

  // ---------------- epilogue: O = acc / l ----------------
  // accl holds rowsum broadcast across the 16 lanes of each group (C layout).
#pragma unroll
  for (int r = 0; r < 8; ++r) {
    const float inv = 1.0f / accl[r];
    const int row = wid * 16 + g * 8 + r;
#pragma unroll
    for (int nt = 0; nt < 8; ++nt)
      Ob[(size_t)row * DH + nt * 16 + n] = acc[nt][r] * inv;
  }
}

extern "C" void kernel_launch(void* const* d_in, const int* in_sizes, int n_in,
                              void* d_out, int out_size, void* d_ws, size_t ws_size,
                              hipStream_t stream) {
  const float* Q = (const float*)d_in[0];
  const float* K = (const float*)d_in[1];
  const float* V = (const float*)d_in[2];
  // d_in[3] is the causal bool mask; causality is applied analytically.
  float* O = (float*)d_out;
  dim3 grid(BATCH * (SQ / BQ));   // 512 blocks
  fa_causal_f16_wmma<<<grid, 256, 0, stream>>>(Q, K, V, O);
}
